// Quantizer_12850542150477
// MI455X (gfx1250) — matrix operand order repaired and checked by reading, not compiled, not run on hardware
//
#include <hip/hip_runtime.h>
#include <hip/hip_bf16.h>

typedef __attribute__((ext_vector_type(16))) __bf16 v16bf;
typedef __attribute__((ext_vector_type(8)))  float  v8f;
typedef __attribute__((ext_vector_type(4)))  float  v4f;

#define KCODES 512
#define DIM    64
#define NROWS  (32 * 8192)
#define NTILES (KCODES / 16)   // 32 code tiles of 16
#define EMA    0.99f

// round-to-nearest-even f32 -> bf16
__device__ __forceinline__ __bf16 f2bf(float f) {
  union { float f; unsigned u; } a; a.f = f;
  unsigned r = a.u + 0x7FFFu + ((a.u >> 16) & 1u);
  union { unsigned short s; __bf16 b; } o; o.s = (unsigned short)(r >> 16);
  return o.b;
}

// ---------------------------------------------------------------------------
// Kernel 1: prepack embedding (f32 row-major -> bf16 in WMMA A-matrix lane
// layout) and compute 0.5*||e_k||^2.
// A layout (16-bit A 16x32, ISA 7.12.2): lane<16 holds M=lane, slots 0..7 =
// K{0..7}, slots 8..15 = K{16..23}; lane>=16 holds M=lane-16, K{8..15,24..31}.
// apack index = ((t*2+q)*32 + lane)*16 + slot, t=code tile, q=k-chunk of 32.
// ---------------------------------------------------------------------------
__global__ void vq_pack_kernel(const float* __restrict__ emb,
                               unsigned short* __restrict__ apack,
                               float* __restrict__ e2half) {
  int tid = blockIdx.x * blockDim.x + threadIdx.x;   // 0..32767
  int s  = tid & 15;
  int l  = (tid >> 4) & 31;
  int q  = (tid >> 9) & 1;
  int t  = tid >> 10;
  int hi = l >> 4;
  int M  = t * 16 + (l & 15);
  int K  = (s < 8) ? (q * 32 + (hi ? 8 : 0)  + s)
                   : (q * 32 + (hi ? 24 : 16) + (s - 8));
  union { __bf16 b; unsigned short u; } cv;
  cv.b = f2bf(emb[M * DIM + K]);
  apack[tid] = cv.u;

  if (tid < KCODES) {
    float acc = 0.f;
    for (int d = 0; d < DIM; ++d) { float e = emb[tid * DIM + d]; acc += e * e; }
    e2half[tid] = 0.5f * acc;
  }
}

__global__ void vq_zero_kernel(float* __restrict__ cnt, float* __restrict__ sums) {
  int tid = blockIdx.x * blockDim.x + threadIdx.x;
  if (tid < KCODES)        cnt[tid]  = 0.f;
  if (tid < KCODES * DIM)  sums[tid] = 0.f;
}

// ---------------------------------------------------------------------------
// Kernel 2: nearest-code assignment via bf16 WMMA + segment accumulation.
// A = 16 codes x 32 k-chunk (LDS, prepacked), B = 32 k-chunk x 16 x-rows.
// dist argmin == argmin_k (0.5*||e_k||^2 - x.e_k)  (||x||^2 constant per row).
// Codebook is staged into LDS with the CDNA5 async-DMA path
// (GLOBAL_LOAD_ASYNC_TO_LDS_B128, ASYNCcnt), overlapping with B-fragment
// construction; completion enforced with s_wait_asynccnt before the barrier.
// ---------------------------------------------------------------------------
__global__ void __launch_bounds__(256)
vq_assign_kernel(const float* __restrict__ x,
                 const unsigned short* __restrict__ apack,
                 const float* __restrict__ e2half,
                 int* __restrict__ idx_out,
                 float* __restrict__ cnt_ws,
                 float* __restrict__ sum_ws) {
  __shared__ __align__(64) unsigned short sA[NTILES * 2 * 32 * 16];  // 64 KB

  // --- async stage: 4096 x 16B, 16 issues per thread, no VGPR round-trip ---
  {
    // AS3 offset == low 32 bits of the generic LDS pointer (aperture rules).
    unsigned lds_base = (unsigned)(size_t)&sA[0];
    unsigned long long gbase = (unsigned long long)(size_t)apack;
#pragma unroll
    for (int j = 0; j < 16; ++j) {
      int i = threadIdx.x + j * 256;           // 16B chunk index, 0..4095
      unsigned           ldst = lds_base + (unsigned)i * 16u;
      unsigned long long gsrc = gbase + (unsigned long long)i * 16ull;
      asm volatile("global_load_async_to_lds_b128 %0, %1, off"
                   :: "v"(ldst), "v"(gsrc) : "memory");
    }
  }

  const int wave = threadIdx.x >> 5;
  const int lane = threadIdx.x & 31;
  const int hi   = lane >> 4;
  const int rowbase = (blockIdx.x * 8 + wave) * 16;
  const int n = rowbase + (lane & 15);
  const float* xrow = x + (size_t)n * DIM;

  // B fragments (overlap with async DMA):
  // lane<16 -> column n, K = 32q+{0..7,16..23} ; lane>=16 -> K = 32q+{8..15,24..31}
  v16bf b[2];
#pragma unroll
  for (int q = 0; q < 2; ++q) {
    const int K0 = q * 32 + hi * 8;
    const float4* p = reinterpret_cast<const float4*>(xrow + K0);
    float4 f0 = p[0], f1 = p[1], f2 = p[4], f3 = p[5];
    float xv[16] = { f0.x, f0.y, f0.z, f0.w, f1.x, f1.y, f1.z, f1.w,
                     f2.x, f2.y, f2.z, f2.w, f3.x, f3.y, f3.z, f3.w };
#pragma unroll
    for (int i = 0; i < 16; ++i) b[q][i] = f2bf(xv[i]);
  }

  asm volatile("s_wait_asynccnt 0x0" ::: "memory");
  __syncthreads();

  float best = 3.4e38f;
  int   bidx = 0;
  const v16bf*  sAv = reinterpret_cast<const v16bf*>(sA);
  const float4* e4  = reinterpret_cast<const float4*>(e2half);

  for (int t = 0; t < NTILES; ++t) {
    v16bf a0 = sAv[(t * 2 + 0) * 32 + lane];
    v16bf a1 = sAv[(t * 2 + 1) * 32 + lane];
    v8f c = {};
    c = __builtin_amdgcn_wmma_f32_16x16x32_bf16(false, a0, false, b[0],
                                                (short)0, c, false, false);
    c = __builtin_amdgcn_wmma_f32_16x16x32_bf16(false, a1, false, b[1],
                                                (short)0, c, false, false);
    // C layout: lane<16 -> (M=j, N=lane); lane>=16 -> (M=j+8, N=lane-16)
    float4 ea = e4[t * 4 + hi * 2];
    float4 eb = e4[t * 4 + hi * 2 + 1];
    float e2v[8] = { ea.x, ea.y, ea.z, ea.w, eb.x, eb.y, eb.z, eb.w };
    const int kbase = t * 16 + hi * 8;
#pragma unroll
    for (int j = 0; j < 8; ++j) {
      float v = e2v[j] - c[j];
      if (v < best) { best = v; bidx = kbase + j; }
    }
  }

  // lane l and l^16 hold the same x-row; combine (tie -> lower code index).
  float ov = __shfl_xor(best, 16, 32);
  int   oi = __shfl_xor(bidx, 16, 32);
  if (ov < best || (ov == best && oi < bidx)) { best = ov; bidx = oi; }

  if (lane < 16) {
    idx_out[n] = bidx;
    atomicAdd(&cnt_ws[bidx], 1.0f);
  }
  const int dstart = hi * 32;  // split the 64-wide scatter across the lane pair
  for (int d = 0; d < 32; ++d)
    atomicAdd(&sum_ws[bidx * DIM + dstart + d], xrow[dstart + d]);
}

// ---------------------------------------------------------------------------
// Kernel 3: EMA update + usage masking -> new codebook.
// (jax.random.randint(key(1)) for dead codes is not reproducible on device;
//  hash stand-in — with ~512 assignments/code expected, usage==1 in practice.)
// ---------------------------------------------------------------------------
__global__ void vq_update_kernel(const float* __restrict__ x,
                                 const float* __restrict__ count_in,
                                 const float* __restrict__ sum_embed_in,
                                 const float* __restrict__ cnt_ws,
                                 const float* __restrict__ sum_ws,
                                 float* __restrict__ emb_new) {
  int tid = blockIdx.x * blockDim.x + threadIdx.x;
  if (tid >= KCODES * DIM) return;
  int k = tid / DIM, d = tid % DIM;
  float c  = cnt_ws[k];
  float cn = count_in[k] + (1.0f - EMA) * (c - count_in[k]);
  bool usage = (cn >= 1.0f);
  float cf = usage ? cn : 1.0f;
  float sn = sum_embed_in[tid] + (1.0f - EMA) * (sum_ws[tid] - sum_embed_in[tid]);
  float en;
  if (usage) {
    en = sn / cf;
  } else {
    unsigned r = ((unsigned)k * 2654435761u) % (unsigned)NROWS;
    en = x[(size_t)r * DIM + d];
  }
  emb_new[tid] = en;
}

// ---------------------------------------------------------------------------
// Kernel 4: quantized = x + (e_new[idx] - x)  (matches the reference's exact
// straight-through rounding). Output is write-once 67 MB -> non-temporal
// stores so it does not evict the L2-resident codebook tensors.
// ---------------------------------------------------------------------------
__global__ void vq_gather_kernel(const int* __restrict__ idx,
                                 const float* __restrict__ emb_new,
                                 const float* __restrict__ x,
                                 float* __restrict__ out) {
  int tid = blockIdx.x * blockDim.x + threadIdx.x;   // NROWS * 16
  int n = tid >> 4, d4 = tid & 15;
  if (n >= NROWS) return;
  const v4f* e  = reinterpret_cast<const v4f*>(emb_new);
  const v4f* xi = reinterpret_cast<const v4f*>(x);
  v4f*       o  = reinterpret_cast<v4f*>(out);
  v4f q  = e[(size_t)idx[n] * 16 + d4];
  v4f xv = xi[(size_t)n * 16 + d4];
  v4f r  = xv + (q - xv);
  __builtin_nontemporal_store(r, &o[(size_t)n * 16 + d4]);
}

extern "C" void kernel_launch(void* const* d_in, const int* in_sizes, int n_in,
                              void* d_out, int out_size, void* d_ws, size_t ws_size,
                              hipStream_t stream) {
  const float* x         = (const float*)d_in[0];   // (32,8192,64) f32
  const float* emb       = (const float*)d_in[1];   // (512,64) f32
  const float* count     = (const float*)d_in[2];   // (512,)  f32
  const float* sum_embed = (const float*)d_in[3];   // (512,64) f32
  float* out = (float*)d_out;                       // (32,8192,64) f32

  char* ws = (char*)d_ws;
  unsigned short* apack  = (unsigned short*)(ws);                       // 64 KB
  float* e2half  = (float*)(ws + 65536);                                //  2 KB
  float* cnt_ws  = (float*)(ws + 65536 + 2048);                         //  2 KB
  float* sum_ws  = (float*)(ws + 65536 + 4096);                         // 128 KB
  float* emb_new = (float*)(ws + 65536 + 4096 + 131072);                // 128 KB
  int*   idx_ws  = (int*)  (ws + 65536 + 4096 + 262144);                //  1 MB

  vq_pack_kernel  <<<128, 256, 0, stream>>>(emb, apack, e2half);
  vq_zero_kernel  <<<128, 256, 0, stream>>>(cnt_ws, sum_ws);
  vq_assign_kernel<<<NROWS / 128, 256, 0, stream>>>(x, apack, e2half,
                                                    idx_ws, cnt_ws, sum_ws);
  vq_update_kernel<<<(KCODES * DIM) / 256, 256, 0, stream>>>(x, count, sum_embed,
                                                             cnt_ws, sum_ws, emb_new);
  vq_gather_kernel<<<(NROWS * 16) / 256, 256, 0, stream>>>(idx_ws, emb_new, x, out);
}
